// ContextEncoderModel_84808424227506
// MI455X (gfx1250) — matrix operand order repaired and checked by reading, not compiled, not run on hardware
//
#include <hip/hip_runtime.h>
#include <stdint.h>
#include <stddef.h>

// ---------------------------------------------------------------------------
// Problem constants (from the reference)
// ---------------------------------------------------------------------------
#define B_    256
#define T_    128
#define D_    300
#define DP_   320      // D padded to multiple of 32 for WMMA K-chunks
#define H_    512
#define G4_   2048     // 4*H
#define OUT_  200
#define OUTP_ 208      // OUT padded to multiple of 16
#define CTXW_ 1024     // 2*H

typedef __attribute__((ext_vector_type(16))) __bf16       v16bf;
typedef __attribute__((ext_vector_type(8)))  float        v8f;
typedef __attribute__((ext_vector_type(4)))  unsigned int v4u;

struct Frag2 { v4u lo, hi; };

// Build a 16-element bf16 WMMA fragment from two 16-byte global loads.
__device__ __forceinline__ v16bf load_frag(const __bf16* p0, const __bf16* p1) {
  Frag2 f;
  f.lo = *(const v4u*)p0;
  f.hi = *(const v4u*)p1;
  return __builtin_bit_cast(v16bf, f);
}

__device__ __forceinline__ __bf16 f2bf(float f) {
  unsigned u = __builtin_bit_cast(unsigned, f);
  u = (u + 0x7FFFu + ((u >> 16) & 1u)) >> 16;   // round-to-nearest-even
  unsigned short s = (unsigned short)u;
  return __builtin_bit_cast(__bf16, s);
}

__device__ __forceinline__ float sigf(float x)   { return 1.0f / (1.0f + __expf(-x)); }
__device__ __forceinline__ float tanh_f(float x) { return 2.0f * sigf(2.0f * x) - 1.0f; }

#define WMMA_BF16(A, Bf, C) \
  __builtin_amdgcn_wmma_f32_16x16x32_bf16(false, (A), false, (Bf), (short)0, (C), false, false)

// ---------------------------------------------------------------------------
// Prep: f32 embeddings [B,T,300] -> bf16 [B,T,320] (zero-padded K tail)
// ---------------------------------------------------------------------------
__global__ void convert_x_kernel(const float* __restrict__ xL,
                                 const float* __restrict__ xR,
                                 __bf16* __restrict__ oL,
                                 __bf16* __restrict__ oR) {
  size_t idx = (size_t)blockIdx.x * blockDim.x + threadIdx.x;
  const size_t total = (size_t)B_ * T_ * DP_;
  if (idx >= total) return;
  int    k   = (int)(idx % DP_);
  size_t row = idx / DP_;
  const float* x = blockIdx.y ? xR : xL;
  __bf16*      o = blockIdx.y ? oR : oL;
  float v = (k < D_) ? x[row * D_ + k] : 0.0f;
  o[idx] = f2bf(v);
}

// ---------------------------------------------------------------------------
// Prep: LSTM weight [(in+H), 4H] f32  ->  transposed + gate-interleaved bf16.
// Permuted row p = hb*64 + g*16 + hi  <->  original column g*512 + hb*16 + hi,
// so one wave's 4 accumulators are the i/j/f/o gates of the same 16 h-units.
// Wxt: [2048, Kxpad] (K zero-padded), Wht: [2048, 512], bperm: [2048] f32.
// ---------------------------------------------------------------------------
__global__ void convert_lstm_w_kernel(const float* __restrict__ W,
                                      const float* __restrict__ b,
                                      int in_dim, int Kxpad,
                                      __bf16* __restrict__ Wxt,
                                      __bf16* __restrict__ Wht,
                                      float* __restrict__ bperm) {
  int p = blockIdx.x * blockDim.x + threadIdx.x;   // 0..2047
  if (p >= G4_) return;
  int hb = p >> 6, g = (p >> 4) & 3, hi = p & 15;
  int col = g * H_ + hb * 16 + hi;
  for (int k = 0; k < Kxpad; ++k)
    Wxt[(size_t)p * Kxpad + k] = (k < in_dim) ? f2bf(W[(size_t)k * G4_ + col]) : f2bf(0.0f);
  for (int k = 0; k < H_; ++k)
    Wht[(size_t)p * H_ + k] = f2bf(W[(size_t)(in_dim + k) * G4_ + col]);
  bperm[p] = b[col];
}

// trans_W [1024, 200] f32 -> transposed bf16 [208, 1024] (zero-padded rows)
__global__ void convert_transw_kernel(const float* __restrict__ W,
                                      __bf16* __restrict__ Wt) {
  int p = blockIdx.x * blockDim.x + threadIdx.x;
  if (p >= OUTP_) return;
  for (int k = 0; k < CTXW_; ++k)
    Wt[(size_t)p * CTXW_ + k] = (p < OUT_) ? f2bf(W[(size_t)k * OUT_ + p]) : f2bf(0.0f);
}

// zero c (f32) and both h double-buffers (bf16) for both stacks
__global__ void zero_states_kernel(float* cL, float* cR,
                                   __bf16* hAL, __bf16* hAR,
                                   __bf16* hBL, __bf16* hBR) {
  int i = blockIdx.x * blockDim.x + threadIdx.x;
  if (i >= B_ * H_) return;
  cL[i] = 0.0f; cR[i] = 0.0f;
  __bf16 z = f2bf(0.0f);
  hAL[i] = z; hAR[i] = z; hBL[i] = z; hBR[i] = z;
}

// ---------------------------------------------------------------------------
// One LSTM timestep for both stacks. Block = 128 threads = 4 waves;
// each wave owns tile (mb in 0..15, hb in 0..31): 16 batch rows x 16 h-units,
// accumulating all four gates via 4 WMMAs per K-chunk of 32.
// grid = (128, 1, 2)  [z: 0=left, 1=right]
// ---------------------------------------------------------------------------
__global__ __launch_bounds__(128)
void lstm_step_kernel(const __bf16* __restrict__ xL, const __bf16* __restrict__ xR,
                      int Kx, int t,
                      const __bf16* __restrict__ WxtL, const __bf16* __restrict__ WxtR,
                      const __bf16* __restrict__ WhtL, const __bf16* __restrict__ WhtR,
                      const float* __restrict__ bpL, const float* __restrict__ bpR,
                      float* __restrict__ cStL, float* __restrict__ cStR,
                      const __bf16* __restrict__ hInL, const __bf16* __restrict__ hInR,
                      __bf16* __restrict__ hOutL, __bf16* __restrict__ hOutR,
                      __bf16* __restrict__ seqL, __bf16* __restrict__ seqR,
                      __bf16* __restrict__ lastL, __bf16* __restrict__ lastR,
                      const int* __restrict__ lenL, const int* __restrict__ lenR) {
  const int stack = blockIdx.z;
  const __bf16* x    = stack ? xR    : xL;
  const __bf16* Wxt  = stack ? WxtR  : WxtL;
  const __bf16* Wht  = stack ? WhtR  : WhtL;
  const float*  bp   = stack ? bpR   : bpL;
  float*        cSt  = stack ? cStR  : cStL;
  const __bf16* hIn  = stack ? hInR  : hInL;
  __bf16*       hOut = stack ? hOutR : hOutL;
  __bf16*       seq  = stack ? seqR  : seqL;
  __bf16*       last = stack ? lastR : lastL;
  const int*    len  = stack ? lenR  : lenL;

  const int lane  = threadIdx.x & 31;
  const int wave  = threadIdx.x >> 5;
  const int tile  = blockIdx.x * 4 + wave;     // 0..511
  const int mb    = tile >> 5;                 // batch block 0..15
  const int hb    = tile & 31;                 // hidden block 0..31
  const int nlane = lane & 15;
  const int kbA   = (lane & 16) ? 16 : 0;      // A-fragment K phase
  const int kbB   = (lane & 16) ? 16 : 0;      // B-fragment K phase
  const int mhalf = (lane & 16) ? 8  : 0;      // C/D M phase

  v8f acc0 = {}, acc1 = {}, acc2 = {}, acc3 = {};

  // ---- x_t @ Wx ----
  const __bf16* xrow = x + ((size_t)(mb * 16 + nlane) * T_ + t) * Kx;
  const __bf16* w0 = Wxt + (size_t)(hb * 64 + 0 * 16 + nlane) * Kx;
  const __bf16* w1 = Wxt + (size_t)(hb * 64 + 1 * 16 + nlane) * Kx;
  const __bf16* w2 = Wxt + (size_t)(hb * 64 + 2 * 16 + nlane) * Kx;
  const __bf16* w3 = Wxt + (size_t)(hb * 64 + 3 * 16 + nlane) * Kx;
  for (int kc = 0; kc < Kx; kc += 32) {
    v16bf a  = load_frag(xrow + kc + kbA, xrow + kc + kbA + 8);
    v16bf b0 = load_frag(w0 + kc + kbB, w0 + kc + kbB + 8);
    v16bf b1 = load_frag(w1 + kc + kbB, w1 + kc + kbB + 8);
    v16bf b2 = load_frag(w2 + kc + kbB, w2 + kc + kbB + 8);
    v16bf b3 = load_frag(w3 + kc + kbB, w3 + kc + kbB + 8);
    acc0 = WMMA_BF16(a, b0, acc0);
    acc1 = WMMA_BF16(a, b1, acc1);
    acc2 = WMMA_BF16(a, b2, acc2);
    acc3 = WMMA_BF16(a, b3, acc3);
  }

  // ---- h_{t-1} @ Wh ----
  const __bf16* hrow = hIn + (size_t)(mb * 16 + nlane) * H_;
  const __bf16* u0 = Wht + (size_t)(hb * 64 + 0 * 16 + nlane) * H_;
  const __bf16* u1 = Wht + (size_t)(hb * 64 + 1 * 16 + nlane) * H_;
  const __bf16* u2 = Wht + (size_t)(hb * 64 + 2 * 16 + nlane) * H_;
  const __bf16* u3 = Wht + (size_t)(hb * 64 + 3 * 16 + nlane) * H_;
  for (int kc = 0; kc < H_; kc += 32) {
    v16bf a  = load_frag(hrow + kc + kbA, hrow + kc + kbA + 8);
    v16bf b0 = load_frag(u0 + kc + kbB, u0 + kc + kbB + 8);
    v16bf b1 = load_frag(u1 + kc + kbB, u1 + kc + kbB + 8);
    v16bf b2 = load_frag(u2 + kc + kbB, u2 + kc + kbB + 8);
    v16bf b3 = load_frag(u3 + kc + kbB, u3 + kc + kbB + 8);
    acc0 = WMMA_BF16(a, b0, acc0);
    acc1 = WMMA_BF16(a, b1, acc1);
    acc2 = WMMA_BF16(a, b2, acc2);
    acc3 = WMMA_BF16(a, b3, acc3);
  }

  // ---- gates + state update (lane-local: acc0..3 = i,j,f,o of same h-unit) ----
  const int   hcol = hb * 16 + nlane;
  const float bI = bp[hb * 64 + 0 * 16 + nlane];
  const float bJ = bp[hb * 64 + 1 * 16 + nlane];
  const float bF = bp[hb * 64 + 2 * 16 + nlane];
  const float bO = bp[hb * 64 + 3 * 16 + nlane];

#pragma unroll
  for (int r = 0; r < 8; ++r) {
    int   mm = mb * 16 + r + mhalf;
    float gi = acc0[r] + bI;
    float gj = acc1[r] + bJ;
    float gf = acc2[r] + bF;
    float go = acc3[r] + bO;
    float cPrev = cSt[(size_t)mm * H_ + hcol];
    float cNew  = cPrev * sigf(gf + 1.0f) + sigf(gi) * tanh_f(gj);  // FORGET_BIAS=1
    float hNew  = tanh_f(cNew) * sigf(go);
    cSt[(size_t)mm * H_ + hcol] = cNew;
    __bf16 hb16 = f2bf(hNew);
    hOut[(size_t)mm * H_ + hcol] = hb16;
    if (seq)  seq[((size_t)mm * T_ + t) * H_ + hcol] = hb16;
    if (last && (t == len[mm] - 1)) last[(size_t)mm * CTXW_ + hcol] = hb16;
  }
}

// ---------------------------------------------------------------------------
// Final projection: ctx [256,1024] bf16 @ trans_W^T [208,1024] bf16 -> f32 out
// 208 wave-tiles (16 M-blocks x 13 N-blocks), 4 waves/block, grid = 52
// ---------------------------------------------------------------------------
__global__ __launch_bounds__(128)
void final_gemm_kernel(const __bf16* __restrict__ ctx,
                       const __bf16* __restrict__ Wt,
                       float* __restrict__ out) {
  const int lane  = threadIdx.x & 31;
  const int wave  = threadIdx.x >> 5;
  const int tile  = blockIdx.x * 4 + wave;   // 0..207
  const int mb    = tile / 13;
  const int nb    = tile % 13;
  const int nlane = lane & 15;
  const int kbA   = (lane & 16) ? 16 : 0;
  const int kbB   = (lane & 16) ? 16 : 0;
  const int mhalf = (lane & 16) ? 8  : 0;

  const __bf16* arow = ctx + (size_t)(mb * 16 + nlane) * CTXW_;
  const __bf16* brow = Wt  + (size_t)(nb * 16 + nlane) * CTXW_;

  v8f acc = {};
  for (int kc = 0; kc < CTXW_; kc += 32) {
    v16bf a = load_frag(arow + kc + kbA, arow + kc + kbA + 8);
    v16bf b = load_frag(brow + kc + kbB, brow + kc + kbB + 8);
    acc = WMMA_BF16(a, b, acc);
  }

  const int col = nb * 16 + nlane;
  if (col < OUT_) {
#pragma unroll
    for (int r = 0; r < 8; ++r) {
      int mm = mb * 16 + r + mhalf;
      out[(size_t)mm * OUT_ + col] = acc[r];
    }
  }
}

// ---------------------------------------------------------------------------
// Host orchestration
// ---------------------------------------------------------------------------
extern "C" void kernel_launch(void* const* d_in, const int* in_sizes, int n_in,
                              void* d_out, int out_size, void* d_ws, size_t ws_size,
                              hipStream_t stream) {
  const float* left   = (const float*)d_in[0];
  const float* right  = (const float*)d_in[1];
  const int*   lenL   = (const int*)d_in[2];
  const int*   lenR   = (const int*)d_in[3];
  const float* lW0 = (const float*)d_in[4];  const float* lb0 = (const float*)d_in[5];
  const float* lW1 = (const float*)d_in[6];  const float* lb1 = (const float*)d_in[7];
  const float* rW0 = (const float*)d_in[8];  const float* rb0 = (const float*)d_in[9];
  const float* rW1 = (const float*)d_in[10]; const float* rb1 = (const float*)d_in[11];
  const float* transW = (const float*)d_in[12];
  float* out = (float*)d_out;

  // Workspace carve-up (~127 MB total)
  char* p = (char*)d_ws;
  auto carve = [&](size_t bytes) -> void* {
    void* r = (void*)p;
    p += (bytes + 255) & ~(size_t)255;
    return r;
  };
  __bf16* xbL   = (__bf16*)carve((size_t)B_ * T_ * DP_ * 2);
  __bf16* xbR   = (__bf16*)carve((size_t)B_ * T_ * DP_ * 2);
  __bf16* WxtL0 = (__bf16*)carve((size_t)G4_ * DP_ * 2);
  __bf16* WhtL0 = (__bf16*)carve((size_t)G4_ * H_ * 2);
  float*  bpL0  = (float*) carve((size_t)G4_ * 4);
  __bf16* WxtR0 = (__bf16*)carve((size_t)G4_ * DP_ * 2);
  __bf16* WhtR0 = (__bf16*)carve((size_t)G4_ * H_ * 2);
  float*  bpR0  = (float*) carve((size_t)G4_ * 4);
  __bf16* WxtL1 = (__bf16*)carve((size_t)G4_ * H_ * 2);
  __bf16* WhtL1 = (__bf16*)carve((size_t)G4_ * H_ * 2);
  float*  bpL1  = (float*) carve((size_t)G4_ * 4);
  __bf16* WxtR1 = (__bf16*)carve((size_t)G4_ * H_ * 2);
  __bf16* WhtR1 = (__bf16*)carve((size_t)G4_ * H_ * 2);
  float*  bpR1  = (float*) carve((size_t)G4_ * 4);
  __bf16* seqL  = (__bf16*)carve((size_t)B_ * T_ * H_ * 2);
  __bf16* seqR  = (__bf16*)carve((size_t)B_ * T_ * H_ * 2);
  float*  cL    = (float*) carve((size_t)B_ * H_ * 4);
  float*  cR    = (float*) carve((size_t)B_ * H_ * 4);
  __bf16* hAL   = (__bf16*)carve((size_t)B_ * H_ * 2);
  __bf16* hAR   = (__bf16*)carve((size_t)B_ * H_ * 2);
  __bf16* hBL   = (__bf16*)carve((size_t)B_ * H_ * 2);
  __bf16* hBR   = (__bf16*)carve((size_t)B_ * H_ * 2);
  __bf16* ctx   = (__bf16*)carve((size_t)B_ * CTXW_ * 2);
  __bf16* Wt2   = (__bf16*)carve((size_t)OUTP_ * CTXW_ * 2);

  // --- prep ---
  {
    size_t total = (size_t)B_ * T_ * DP_;
    dim3 g((unsigned)((total + 255) / 256), 2, 1);
    convert_x_kernel<<<g, 256, 0, stream>>>(left, right, xbL, xbR);
  }
  convert_lstm_w_kernel<<<8, 256, 0, stream>>>(lW0, lb0, D_, DP_, WxtL0, WhtL0, bpL0);
  convert_lstm_w_kernel<<<8, 256, 0, stream>>>(rW0, rb0, D_, DP_, WxtR0, WhtR0, bpR0);
  convert_lstm_w_kernel<<<8, 256, 0, stream>>>(lW1, lb1, H_, H_,  WxtL1, WhtL1, bpL1);
  convert_lstm_w_kernel<<<8, 256, 0, stream>>>(rW1, rb1, H_, H_,  WxtR1, WhtR1, bpR1);
  convert_transw_kernel<<<1, 256, 0, stream>>>(transW, Wt2);

  dim3 sgrid(128, 1, 2);

  // --- layer 0 (writes full h-sequence; layer-1 input) ---
  zero_states_kernel<<<(B_ * H_) / 256, 256, 0, stream>>>(cL, cR, hAL, hAR, hBL, hBR);
  for (int t = 0; t < T_; ++t) {
    const __bf16* hiL = (t & 1) ? hBL : hAL;
    const __bf16* hiR = (t & 1) ? hBR : hAR;
    __bf16* hoL = (t & 1) ? hAL : hBL;
    __bf16* hoR = (t & 1) ? hAR : hBR;
    lstm_step_kernel<<<sgrid, 128, 0, stream>>>(
        xbL, xbR, DP_, t,
        WxtL0, WxtR0, WhtL0, WhtR0, bpL0, bpR0,
        cL, cR, hiL, hiR, hoL, hoR,
        seqL, seqR, (__bf16*)nullptr, (__bf16*)nullptr, lenL, lenR);
  }

  // --- layer 1 (writes h at t = len-1 into ctx) ---
  zero_states_kernel<<<(B_ * H_) / 256, 256, 0, stream>>>(cL, cR, hAL, hAR, hBL, hBR);
  for (int t = 0; t < T_; ++t) {
    const __bf16* hiL = (t & 1) ? hBL : hAL;
    const __bf16* hiR = (t & 1) ? hBR : hAR;
    __bf16* hoL = (t & 1) ? hAL : hBL;
    __bf16* hoR = (t & 1) ? hAR : hBR;
    lstm_step_kernel<<<sgrid, 128, 0, stream>>>(
        seqL, seqR, H_, t,
        WxtL1, WxtR1, WhtL1, WhtR1, bpL1, bpR1,
        cL, cR, hiL, hiR, hoL, hoR,
        (__bf16*)nullptr, (__bf16*)nullptr, ctx, ctx + H_, lenL, lenR);
  }

  // --- final projection ---
  final_gemm_kernel<<<52, 128, 0, stream>>>(ctx, Wt2, out);
}
